// LSTMcell_76089640616544
// MI455X (gfx1250) — matrix operand order repaired and checked by reading, not compiled, not run on hardware
//
#include <hip/hip_runtime.h>
#include <hip/hip_fp16.h>

// BN-LSTM on MI455X (gfx1250, wave32, WMMA), persistent-kernel version.
//
// Roofline: ~0.72 TFLOP over 512 *sequentially dependent* steps; weights
// (2.75 MB f16) + state (~10 MB) are L2-resident (192 MB), so per-step HBM
// traffic is negligible vs 23.3 TB/s and each step's ~1.4 GFLOP of f16 WMMA
// is ~1 us. Latency of synchronization dominates, so: ONE persistent kernel
// (128 blocks = 1024 waves, one wave per 16x64 gate tile) with only TWO
// device-wide barriers per step (X-slice staging for step t+1 is fused into
// the pointwise phase of step t; the output GEMM overlaps the next step's
// gate GEMM via the barrier schedule).
//
// Codegen notes:
//  * An opaque zero offset (volatile asm) is added to the weight pointers
//    every step. This keeps the loads loop-variant so LICM cannot hoist 80
//    B-fragments out of the t-loop (round 2: >256 VGPRs + scratch spills),
//    while preserving pointer provenance so they stay global_load_b128
//    (round 3 showed "+s" on the pointer itself degrades them to flat_load,
//    which burns DScnt and LDS-path arbitration).
//  * BN stats are deterministic: in-register column sums -> per-mtile
//    partials -> block-level finalize into LDS. No float atomics.
//  * Gate biases cancel under BatchNorm (BN(x+b)=BN(x)) and are omitted.

typedef __attribute__((ext_vector_type(16))) _Float16 v16h;
typedef __attribute__((ext_vector_type(8)))  float    v8f;

#define B_    512
#define T_    512
#define NIN_  128
#define DH_   512
#define NOUT_ 128
#define KCOMB 640    // NIN + DH
#define NG    2048   // 4 * DH, gate order: f, i, o, c
#define EPS_  1e-5f

#define NBLK  128
#define NTHR  256
#define GSIZE (NBLK * NTHR)   // 32768 threads, 1024 waves

// ---- WMMA fragment loaders (ISA 7.12.2 layouts) -------------------------

// A 16x32 f16: lanes 0-15 row M=lane, K={0..7,16..23};
// lanes 16-31 row M=lane-16, K={8..15,24..31}. Two 16B loads.
__device__ __forceinline__ v16h load_a16x32(const _Float16* __restrict__ base, int ld) {
  const int lane = threadIdx.x & 31;
  const int row  = lane & 15;
  const int koff = (lane >> 4) << 3;
  const _Float16* p = base + row * ld + koff;
  union { v16h v; float4 q[2]; } u;
  u.q[0] = *(const float4*)(p);
  u.q[1] = *(const float4*)(p + 16);
  return u.v;
}

// B 32x16 f16 (column per lane): lanes 0-15 col N=lane K={0..7,16..23},
// lanes 16-31 col N=lane-16 K={8..15,24..31}. B column n == row n of row-major W.
__device__ __forceinline__ v16h load_b32x16(const _Float16* __restrict__ Wrow0, int ld) {
  const int lane = threadIdx.x & 31;
  const int col  = lane & 15;
  const int kb   = (lane >> 4) << 3;
  const _Float16* p = Wrow0 + col * ld + kb;
  union { v16h v; float4 q[2]; } u;
  u.q[0] = *(const float4*)(p);
  u.q[1] = *(const float4*)(p + 16);
  return u.v;
}

// ---- device-wide barrier (persistent kernel) ----------------------------

__device__ __forceinline__ void grid_barrier(int* counter, int target) {
  __threadfence();                       // publish this thread's global writes
  __syncthreads();
  if (threadIdx.x == 0) {
    __hip_atomic_fetch_add(counter, 1, __ATOMIC_RELEASE, __HIP_MEMORY_SCOPE_AGENT);
    while (__hip_atomic_load(counter, __ATOMIC_ACQUIRE, __HIP_MEMORY_SCOPE_AGENT) < target)
      __builtin_amdgcn_s_sleep(2);       // back off while spinning
  }
  __syncthreads();
}

// ---- one-time setup kernels ---------------------------------------------

__global__ void pack_gate_weights(const float* __restrict__ Wf, const float* __restrict__ Wi,
                                  const float* __restrict__ Wo, const float* __restrict__ Wc,
                                  _Float16* __restrict__ Wg) {
  int idx = blockIdx.x * blockDim.x + threadIdx.x;   // NG*KCOMB threads
  int n = idx / KCOMB;
  int k = idx - n * KCOMB;
  int gate = n >> 9, r = n & 511;
  const float* W = gate == 0 ? Wf : gate == 1 ? Wi : gate == 2 ? Wo : Wc;
  Wg[idx] = (_Float16)W[r * KCOMB + k];
}

__global__ void pack_out_weights(const float* __restrict__ Wout, _Float16* __restrict__ W) {
  int idx = blockIdx.x * blockDim.x + threadIdx.x;   // NOUT*DH threads
  W[idx] = (_Float16)Wout[idx];
}

__global__ void init_state(float* __restrict__ cstate, _Float16* __restrict__ comb,
                           int* __restrict__ counter) {
  int idx = blockIdx.x * blockDim.x + threadIdx.x;   // B*DH threads
  cstate[idx] = 0.f;
  int b = idx >> 9, j = idx & 511;
  comb[b * KCOMB + NIN_ + j] = (_Float16)0.f;        // h0 = 0
  if (idx == 0) *counter = 0;                        // reset grid barrier
}

// ---- the persistent LSTM kernel -----------------------------------------

__global__ void __launch_bounds__(NTHR, 1)
lstm_persistent(const float* __restrict__ X,
                const _Float16* __restrict__ Wg_in,
                const _Float16* __restrict__ WoutH_in,
                const float* __restrict__ gf, const float* __restrict__ bef,
                const float* __restrict__ gi, const float* __restrict__ bei,
                const float* __restrict__ go, const float* __restrict__ beo,
                const float* __restrict__ gc, const float* __restrict__ bec,
                const float* __restrict__ bout,
                float* __restrict__ gates,          // [512 x 2048]
                float* __restrict__ psum,           // [32 x 2048] per-mtile sums
                float* __restrict__ psq,            // [32 x 2048] per-mtile sum-sq
                _Float16* __restrict__ comb,        // [512 x 640]
                float* __restrict__ cstate,         // [512 x 512]
                float* __restrict__ out,            // [512 x 512 x 128]
                int* __restrict__ counter) {
  __shared__ float sm_mean[128];
  __shared__ float sm_rstd[128];

  const int tid  = threadIdx.x;
  const int gtid = blockIdx.x * NTHR + tid;          // 0 .. 32767
  const int wave = gtid >> 5;                        // 0 .. 1023
  const int lane = tid & 31;

  // fixed gate-GEMM tile ownership (1024 waves == 32 mtiles x 32 ntiles)
  const int mtile = wave >> 5, ntile = wave & 31;
  const int m0 = mtile * 16, n0 = ntile * 64;

  // fixed pointwise patch ownership: 64 rows x 32 hidden-cols per block
  const int j0 = (blockIdx.x & 15) * 32;
  const int b0 = (blockIdx.x >> 4) * 64;

  int round = 0;

  // ---- prologue: stage X slice for t = 0 ----
  for (int idx = gtid; idx < B_ * NIN_; idx += GSIZE) {
    int b = idx >> 7, k = idx & 127;
    comb[b * KCOMB + k] = (_Float16)X[(b * T_) * NIN_ + k];
  }
  grid_barrier(counter, ++round * NBLK);

  for (int t = 0; t < T_; ++t) {
    // Opaque zero offset: keeps weight loads loop-variant (no LICM hoist /
    // spill) while preserving global address-space provenance.
    long zofs = 0;
    asm volatile("" : "+s"(zofs));
    const _Float16* Wg    = Wg_in + zofs;
    const _Float16* WoutH = WoutH_in + zofs;

    // ---- phase B: gate GEMM (wave-per-16x64 tile) + stat partials ----
    {
      v8f acc[4] = {};
      const _Float16* arow = comb + m0 * KCOMB;
      for (int kk = 0; kk < KCOMB; kk += 32) {
        if (kk + 32 < KCOMB)
          __builtin_prefetch(arow + kk + 32, 0, 3);  // near-cache prefetch
        v16h a = load_a16x32(arow + kk, KCOMB);
#pragma unroll
        for (int j = 0; j < 4; ++j) {
          v16h b = load_b32x16(Wg + (n0 + j * 16) * KCOMB + kk, KCOMB);
          acc[j] = __builtin_amdgcn_wmma_f32_16x16x32_f16(false, a, false, b,
                                                          (short)0, acc[j], false, false);
        }
      }
      const int col = lane & 15;
      const int rb  = (lane >> 4) << 3;
#pragma unroll
      for (int j = 0; j < 4; ++j) {
        float s = 0.f, ss = 0.f;
#pragma unroll
        for (int v = 0; v < 8; ++v) {
          float x = acc[j][v];
          gates[(m0 + rb + v) * NG + n0 + j * 16 + col] = x;
          s += x; ss += x * x;
        }
        // lane l and l^16 hold the same column's two 8-row halves
        s  += __shfl_xor(s, 16);
        ss += __shfl_xor(ss, 16);
        if (lane < 16) {                             // deterministic partials
          psum[mtile * NG + n0 + j * 16 + col] = s;
          psq [mtile * NG + n0 + j * 16 + col] = ss;
        }
      }
    }
    grid_barrier(counter, ++round * NBLK);

    // ---- phase C: finalize BN stats + pointwise update + stage X(t+1) ----
    if (tid < 128) {                                  // 4 gates x 32 cols
      int gate = tid >> 5, jj = tid & 31;
      int col = gate * 512 + j0 + jj;
      float s = 0.f, ss = 0.f;
#pragma unroll 8
      for (int m = 0; m < 32; ++m) {
        s  += psum[m * NG + col];
        ss += psq [m * NG + col];
      }
      float mean = s * (1.f / B_);
      float var  = ss * (1.f / B_) - mean * mean;     // biased, as reference
      sm_mean[tid] = mean;
      sm_rstd[tid] = rsqrtf(var + EPS_);
    }
    __syncthreads();
#pragma unroll
    for (int i = 0; i < 8; ++i) {                     // 2048 elems / 256 thr
      int local = tid + i * NTHR;
      int jj = local & 31, j = j0 + jj;
      int b  = b0 + (local >> 5);
      const float* g = gates + b * NG;
      float nf = gf[j] * (g[j]        - sm_mean[jj])      * sm_rstd[jj]      + bef[j];
      float ni = gi[j] * (g[512 + j]  - sm_mean[32 + jj]) * sm_rstd[32 + jj] + bei[j];
      float no = go[j] * (g[1024 + j] - sm_mean[64 + jj]) * sm_rstd[64 + jj] + beo[j];
      float nc = gc[j] * (g[1536 + j] - sm_mean[96 + jj]) * sm_rstd[96 + jj] + bec[j];
      float f  = 1.f / (1.f + __expf(-nf));
      float ig = 1.f / (1.f + __expf(-ni));
      float o  = 1.f / (1.f + __expf(-no));
      float cd = tanhf(nc);
      int sidx = b * DH_ + j;
      float c = f * cstate[sidx] + ig * cd;
      cstate[sidx] = c;
      float h = o * tanhf(c);
      comb[b * KCOMB + NIN_ + j] = (_Float16)h;       // next-step A & out-GEMM A
    }
    // stage the X slice for step t+1 (conflicts only with B(t), done above)
    if (t + 1 < T_) {
      for (int idx = gtid; idx < B_ * NIN_; idx += GSIZE) {
        int b = idx >> 7, k = idx & 127;
        comb[b * KCOMB + k] = (_Float16)X[(b * T_ + (t + 1)) * NIN_ + k];
      }
    }
    grid_barrier(counter, ++round * NBLK);

    // ---- phase D: y[:, t, :] = tanh(h * Wout^T + bout) (256 waves, 16x16) ----
    // D(t) reads comb's h region; it completes before these waves signal the
    // next step's post-GEMM barrier, which precedes the C(t+1) overwrite.
    if (wave < 256) {
      const int dm0 = (wave >> 3) * 16;               // 32 mtiles
      const int dn0 = (wave & 7) * 16;                // 8 ntiles of 16
      const _Float16* h = comb + NIN_;
      v8f acc = {};
      for (int kk = 0; kk < DH_; kk += 32) {
        v16h a = load_a16x32(h + dm0 * KCOMB + kk, KCOMB);
        v16h b = load_b32x16(WoutH + dn0 * DH_ + kk, DH_);
        acc = __builtin_amdgcn_wmma_f32_16x16x32_f16(false, a, false, b,
                                                     (short)0, acc, false, false);
      }
      const int col = lane & 15;
      const int rb  = (lane >> 4) << 3;
#pragma unroll
      for (int v = 0; v < 8; ++v) {
        int n = dn0 + col;
        int m = dm0 + rb + v;
        out[(m * T_ + t) * NOUT_ + n] = tanhf(acc[v] + bout[n]);
      }
    }
  }
}

// ---- launcher -----------------------------------------------------------

extern "C" void kernel_launch(void* const* d_in, const int* in_sizes, int n_in,
                              void* d_out, int out_size, void* d_ws, size_t ws_size,
                              hipStream_t stream) {
  (void)in_sizes; (void)n_in; (void)out_size; (void)ws_size;
  const float* X    = (const float*)d_in[0];
  const float* Wf   = (const float*)d_in[1];
  const float* gf   = (const float*)d_in[3];
  const float* bef  = (const float*)d_in[4];
  const float* Wi   = (const float*)d_in[5];
  const float* gi   = (const float*)d_in[7];
  const float* bei  = (const float*)d_in[8];
  const float* Wo   = (const float*)d_in[9];
  const float* go   = (const float*)d_in[11];
  const float* beo  = (const float*)d_in[12];
  const float* Wc   = (const float*)d_in[13];
  const float* gc   = (const float*)d_in[15];
  const float* bec  = (const float*)d_in[16];
  const float* Wout = (const float*)d_in[17];
  const float* bout = (const float*)d_in[18];
  float* out = (float*)d_out;

  // workspace carve-up (~9.2 MB total; all chunks 256B-aligned)
  char* ws = (char*)d_ws;
  _Float16* Wg   = (_Float16*)ws; ws += (size_t)NG * KCOMB * sizeof(_Float16);
  _Float16* WoH  = (_Float16*)ws; ws += (size_t)NOUT_ * DH_ * sizeof(_Float16);
  _Float16* comb = (_Float16*)ws; ws += (size_t)B_ * KCOMB * sizeof(_Float16);
  float* gates   = (float*)ws;    ws += (size_t)B_ * NG * sizeof(float);
  float* psum    = (float*)ws;    ws += (size_t)32 * NG * sizeof(float);
  float* psq     = (float*)ws;    ws += (size_t)32 * NG * sizeof(float);
  float* cstate  = (float*)ws;    ws += (size_t)B_ * DH_ * sizeof(float);
  int* counter   = (int*)ws;      ws += 256;

  pack_gate_weights<<<(NG * KCOMB) / NTHR, NTHR, 0, stream>>>(Wf, Wi, Wo, Wc, Wg);
  pack_out_weights<<<(NOUT_ * DH_) / NTHR, NTHR, 0, stream>>>(Wout, WoH);
  init_state<<<(B_ * DH_) / NTHR, NTHR, 0, stream>>>(cstate, comb, counter);

  lstm_persistent<<<NBLK, NTHR, 0, stream>>>(X, Wg, WoH,
                                             gf, bef, gi, bei, go, beo, gc, bec,
                                             bout, gates, psum, psq,
                                             comb, cstate, out, counter);
}
